// FORTZAModel_65781719106122
// MI455X (gfx1250) — matrix operand order repaired and checked by reading, not compiled, not run on hardware
//
#include <hip/hip_runtime.h>

typedef __attribute__((ext_vector_type(16))) _Float16 v16h;
typedef __attribute__((ext_vector_type(8)))  float    v8f;
typedef __attribute__((ext_vector_type(4)))  unsigned u32x4;
typedef __attribute__((ext_vector_type(8)))  unsigned u32x8;

namespace {
constexpr int B_   = 512;
constexpr int T_IN = 256;
constexpr int T_OUT= 128;
constexpr int V_   = 256;
constexpr int N_   = 1024;
constexpr int EH   = 64;
constexpr int DH   = 128;
constexpr int SD   = 32;
constexpr int GE   = 4 * EH;   // 256 encoder gate width
constexpr int GD   = 4 * DH;   // 512 decoder gate width
}

__device__ __forceinline__ float sigmf(float x) { return 1.0f / (1.0f + __expf(-x)); }

// D = A(16x32 f16) * B(32x16 f16) + C(16x16 f32), wave32 WMMA.
__device__ __forceinline__ v8f wmma16(v16h a, v16h b, v8f c) {
  return __builtin_amdgcn_wmma_f32_16x16x32_f16(
      /*neg_a=*/false, a, /*neg_b=*/false, b,
      /*c_mod=*/(short)0, c, /*reuse_a=*/false, /*reuse_b=*/false);
}

// ---------------------------------------------------------------------------
// TDM gather: DMA 16 index-selected rows of a 2D f32 table (row_elems wide,
// n_rows tall) into contiguous LDS at lds_off.  D# per CDNA5 ISA ch.8:
//   group0: count=1 | gather_mode(b31) | idx16(b30=0), lds_addr, gaddr, type=2
//   group1: data_size=4B(b17:16), tensor_dim0/1, tile_dim0, tile_dim1=16, stride0
//   group2/3: 16 packed 16-bit row indices
// Issued once per wave (EXEC ignored); tracked with TENSORcnt.
__device__ __forceinline__ void tdm_gather16(unsigned lds_off, const void* gbase,
                                             unsigned row_elems, unsigned n_rows,
                                             const unsigned* idxp) {
  unsigned long long ga = (unsigned long long)(uintptr_t)gbase;
  u32x4 g0; u32x8 g1; u32x4 g2; u32x4 g3;
  g0[0] = 0x80000001u;                                        // count=1, gather_mode=1
  g0[1] = lds_off;                                            // LDS byte address
  g0[2] = (unsigned)ga;                                       // global_addr[31:0]
  g0[3] = (unsigned)((ga >> 32) & 0x1FFFFFFu) | 0x80000000u;  // addr[56:32] | type=2
  g1[0] = 2u << 16;                                           // data_size = 4 bytes
  g1[1] = (row_elems & 0xFFFFu) << 16;                        // tensor_dim0 lo16
  g1[2] = ((row_elems >> 16) & 0xFFFFu) | ((n_rows & 0xFFFFu) << 16); // dim0 hi | dim1 lo
  g1[3] = ((n_rows >> 16) & 0xFFFFu) | ((row_elems & 0xFFFFu) << 16); // dim1 hi | tile_dim0
  g1[4] = 16u;                                                // tile_dim1 = #valid indices
  g1[5] = row_elems;                                          // tensor_dim0_stride lo32
  g1[6] = 0u;
  g1[7] = 0u;
  g2[0] = idxp[0]; g2[1] = idxp[1]; g2[2] = idxp[2]; g2[3] = idxp[3];
  g3[0] = idxp[4]; g3[1] = idxp[5]; g3[2] = idxp[6]; g3[3] = idxp[7];
  asm volatile("tensor_load_to_lds %0, %1, %2, %3"
               :: "s"(g0), "s"(g1), "s"(g2), "s"(g3)
               : "memory");
}

// ---------------------------------------------------------------------------
// One wave per one-hot row of 256 floats: recover the hot index.
__global__ void k_argmax(const float* __restrict__ oh, int* __restrict__ idx, int rows) {
  int row  = blockIdx.x * 8 + (threadIdx.x >> 5);
  int lane = threadIdx.x & 31;
  if (row >= rows) return;
  const float* p = oh + (size_t)row * V_;
  int my = 0;
#pragma unroll
  for (int i = 0; i < 8; ++i) {
    float v = p[lane * 8 + i];
    if (v > 0.5f) my = lane * 8 + i;
  }
#pragma unroll
  for (int off = 16; off > 0; off >>= 1) {
    int o = __shfl_xor(my, off, 32);
    my = (o > my) ? o : my;
  }
  if (lane == 0) idx[row] = my;
}

// ---------------------------------------------------------------------------
// dst[v][g] = src[g*ld + v] (+ bias[g])  -- token-major gather tables (f32).
__global__ void k_packT(const float* __restrict__ src, int ld,
                        const float* __restrict__ bias, int G, float* __restrict__ dst) {
  int t = blockIdx.x * 256 + threadIdx.x;
  if (t >= V_ * G) return;
  int v = t / G, g = t % G;
  float x = src[(size_t)g * ld + v];
  if (bias) x += bias[g];
  dst[(size_t)v * G + g] = x;
}

// ---------------------------------------------------------------------------
// Pack Bmat[k][n] = src[n*ld + off + k] into WMMA B-operand swizzle (f16):
// chunk (kc = k/32, nt = n/16); lane = n%16 + 16*((k%16)/8); e = k%8 + 8*((k%32)/16).
__global__ void k_packB(const float* __restrict__ src, int ld, int off, int K, int N,
                        _Float16* __restrict__ dst) {
  int t = blockIdx.x * 256 + threadIdx.x;
  if (t >= K * N) return;
  int k = t / N, n = t % N;
  int kc = k >> 5, kk = k & 31, nt = n >> 4;
  int lane = (n & 15) | (((kk >> 3) & 1) << 4);
  int e    = (kk & 7) | (((kk >> 4) & 1) << 3);
  int NT = N >> 4;
  dst[((((size_t)kc * NT + nt) * 32 + lane) << 4) + e] = (_Float16)src[(size_t)n * ld + off + k];
}

// Pack A[m][k] = src[m*K + k] into WMMA A-operand swizzle (f16).
__global__ void k_packA(const float* __restrict__ src, int M, int K, _Float16* __restrict__ dst) {
  int t = blockIdx.x * 256 + threadIdx.x;
  if (t >= M * K) return;
  int m = t / K, k = t % K;
  int mt = m >> 4, row = m & 15, kc = k >> 5, kk = k & 31;
  int lane = row | (((kk >> 3) & 1) << 4);
  int e    = (kk & 7) | (((kk >> 4) & 1) << 3);
  int KC = K >> 5;
  dst[((((size_t)mt * KC + kc) * 32 + lane) << 4) + e] = (_Float16)src[(size_t)m * K + k];
}

// ---------------------------------------------------------------------------
// gx_const = zia(512x1024) @ Wz.T(1024x256) + enc_b.
__global__ void k_gxconst(const _Float16* __restrict__ A, const _Float16* __restrict__ Bm,
                          const float* __restrict__ bias, float* __restrict__ out) {
  int mt = blockIdx.x;
  int wave = threadIdx.x >> 5, lane = threadIdx.x & 31;
  v8f acc[2] = {};
  for (int kc = 0; kc < 32; ++kc) {
    v16h a = *(const v16h*)(A + ((((size_t)mt * 32 + kc) * 32 + lane) << 4));
#pragma unroll
    for (int j = 0; j < 2; ++j) {
      int nt = wave * 2 + j;
      v16h b = *(const v16h*)(Bm + ((((size_t)kc * 16 + nt) * 32 + lane) << 4));
      acc[j] = wmma16(a, b, acc[j]);
    }
  }
  int n = lane & 15, hi = lane >> 4;
#pragma unroll
  for (int j = 0; j < 2; ++j) {
    int g = (wave * 2 + j) * 16 + n;
#pragma unroll
    for (int r = 0; r < 8; ++r) {
      int bb = mt * 16 + r + 8 * hi;
      out[(size_t)bb * GE + g] = acc[j][r] + bias[g];
    }
  }
}

// ---------------------------------------------------------------------------
// Encoder LSTM scan: 32 blocks x 16 batch rows, 256 steps.  Per step the TDM
// gathers the 16 token rows of WxT straight into the LDS gates buffer while
// WMMA consumes register-resident Whh tiles; gx_const is a preloaded LDS term.
__global__ void k_enc_scan(const int* __restrict__ inp_idx, const float* __restrict__ WxT,
                           const float* __restrict__ gxc, const _Float16* __restrict__ WhhB,
                           float* __restrict__ h_out) {
  __shared__ __align__(32) _Float16 hswz[2][32][16];   // h in A-operand swizzle
  __shared__ float gates[16][GE];
  __shared__ float gxcl[16][GE];
  __shared__ float cbuf[16][EH];
  int btile = blockIdx.x, tid = threadIdx.x;
  int wave = tid >> 5, lane = tid & 31;
  int b0 = btile * 16;

#pragma unroll
  for (int i = 0; i < 4; ++i) {                  // zero h (1024 halves), c (1024 f32)
    ((_Float16*)hswz)[tid + i * 256] = (_Float16)0.0f;
    int idx = tid + i * 256;
    cbuf[idx >> 6][idx & 63] = 0.0f;
  }
#pragma unroll
  for (int i = 0; i < 16; ++i)                   // t-invariant input-side gates
    gxcl[i][tid] = gxc[(size_t)(b0 + i) * GE + tid];

  v16h Breg[2][2];                               // Whh^T tiles, resident across scan
#pragma unroll
  for (int j = 0; j < 2; ++j)
#pragma unroll
    for (int kc = 0; kc < 2; ++kc)
      Breg[j][kc] = *(const v16h*)(WhhB + ((((size_t)kc * 16 + (wave * 2 + j)) * 32 + lane) << 4));
  __syncthreads();

  unsigned gates_lds = (unsigned)(uintptr_t)(void*)&gates[0][0];

  for (int t = 0; t < T_IN; ++t) {
    if (wave == 0) {                             // DMA gather of WxT[token] rows
      unsigned idxp[8];
#pragma unroll
      for (int p = 0; p < 8; ++p) {
        unsigned i0 = (unsigned)inp_idx[(size_t)(b0 + 2 * p) * T_IN + t];
        unsigned i1 = (unsigned)inp_idx[(size_t)(b0 + 2 * p + 1) * T_IN + t];
        idxp[p] = (i0 & 0xFFFFu) | (i1 << 16);
      }
      if (t + 1 < T_IN) __builtin_prefetch(inp_idx + (size_t)b0 * T_IN + t + 1, 0, 2);
      tdm_gather16(gates_lds, WxT, GE, V_, idxp);
      __builtin_amdgcn_s_wait_tensorcnt(0);
    }
    __syncthreads();

    v16h a0 = *(const v16h*)(&hswz[0][lane][0]);
    v16h a1 = *(const v16h*)(&hswz[1][lane][0]);
    v8f acc[2] = {};
#pragma unroll
    for (int j = 0; j < 2; ++j) {
      acc[j] = wmma16(a0, Breg[j][0], acc[j]);
      acc[j] = wmma16(a1, Breg[j][1], acc[j]);
    }
    int n = lane & 15, hi = lane >> 4;
#pragma unroll
    for (int j = 0; j < 2; ++j) {
      int g = (wave * 2 + j) * 16 + n;
#pragma unroll
      for (int r = 0; r < 8; ++r) {
        int m = r + 8 * hi;
        gates[m][g] += acc[j][r] + gxcl[m][g];
      }
    }
    __syncthreads();

    // gate nonlinearity + state update (gate order i,f,g,o)
#pragma unroll
    for (int i = 0; i < 4; ++i) {
      int idx = tid + i * 256;
      int m = idx >> 6, jj = idx & 63;
      float ig = gates[m][jj], fg = gates[m][EH + jj];
      float gg = gates[m][2 * EH + jj], og = gates[m][3 * EH + jj];
      float c2 = sigmf(fg) * cbuf[m][jj] + sigmf(ig) * tanhf(gg);
      float h2 = sigmf(og) * tanhf(c2);
      cbuf[m][jj] = c2;
      int kc = jj >> 5, kk = jj & 31;
      hswz[kc][m | (((kk >> 3) & 1) << 4)][(kk & 7) | (((kk >> 4) & 1) << 3)] = (_Float16)h2;
      if (t == T_IN - 1) h_out[(size_t)(b0 + m) * EH + jj] = h2;
    }
    __syncthreads();
  }
}

// ---------------------------------------------------------------------------
// fortza gate/update (tiny) -> fortza_new written straight to d_out tail.
__global__ void k_sg_fortza(const float* __restrict__ h_enc, const float* __restrict__ fortza,
                            const float* __restrict__ sgW, const float* __restrict__ sgb,
                            float* __restrict__ fnew) {
  int t = blockIdx.x * 256 + threadIdx.x;
  if (t >= B_ * SD) return;
  int b = t >> 5, s = t & 31;
  float d1 = sgb[s], d2 = sgb[SD + s];
  const float* w1 = sgW + (size_t)s * (EH + SD);
  const float* w2 = sgW + (size_t)(SD + s) * (EH + SD);
  for (int k = 0; k < EH; ++k) { float hv = h_enc[(size_t)b * EH + k]; d1 += hv * w1[k]; d2 += hv * w2[k]; }
  for (int k = 0; k < SD; ++k) { float fv = fortza[(size_t)b * SD + k]; d1 += fv * w1[EH + k]; d2 += fv * w2[EH + k]; }
  float gate = sigmf(d1), upd = tanhf(d2);
  fnew[t] = gate * fortza[(size_t)b * SD + s] + (1.0f - gate) * upd;
}

// dh0 = tanh(concat(h_enc, zia) @ Wc_W.T + Wc_b)   (512x128, K=1088; small)
__global__ void k_dh0(const float* __restrict__ h_enc, const float* __restrict__ zia,
                      const float* __restrict__ WcW, const float* __restrict__ Wcb,
                      float* __restrict__ dh0) {
  int t = blockIdx.x * 256 + threadIdx.x;
  if (t >= B_ * DH) return;
  int b = t >> 7, j = t & 127;
  const float* w = WcW + (size_t)j * (EH + N_);
  float acc = Wcb[j];
  for (int k = 0; k < EH; ++k) acc += h_enc[(size_t)b * EH + k] * w[k];
  for (int k = 0; k < N_; ++k) acc += zia[(size_t)b * N_ + k] * w[EH + k];
  dh0[t] = tanhf(acc);
}

// f_logit[b][v] = fortza_new @ Wf.T + out_b   (per-batch logit bias)
__global__ void k_flogit(const float* __restrict__ fnew, const float* __restrict__ outW,
                         const float* __restrict__ outb, float* __restrict__ flog) {
  int t = blockIdx.x * 256 + threadIdx.x;
  if (t >= B_ * V_) return;
  int b = t >> 8, v = t & 255;
  float acc = outb[v];
  const float* w = outW + (size_t)v * (DH + SD) + DH;
  for (int s = 0; s < SD; ++s) acc += fnew[(size_t)b * SD + s] * w[s];
  flog[t] = acc;
}

// ---------------------------------------------------------------------------
// Decoder LSTM scan: 32 blocks x 16 rows, 128 steps, DH=128 / 512 gates.
// TDM gathers gxdecT[token] rows (bias folded in) into the gates buffer each
// step; 16 resident B-tiles per wave; h written to LDS and A-swizzled dhs.
__global__ void k_dec_scan(const int* __restrict__ tgt_idx, const float* __restrict__ gxdecT,
                           const float* __restrict__ dh0, const _Float16* __restrict__ WhhB,
                           _Float16* __restrict__ dhsA) {
  __shared__ __align__(32) _Float16 hswz[4][32][16];
  __shared__ float gates[16][GD];
  __shared__ float cbuf[16][DH];
  int btile = blockIdx.x, tid = threadIdx.x;
  int wave = tid >> 5, lane = tid & 31;
  int b0 = btile * 16;

#pragma unroll
  for (int i = 0; i < 8; ++i) {                  // h <- dh0, c <- 0
    int idx = tid + i * 256;
    int m = idx >> 7, jj = idx & 127;
    float h = dh0[(size_t)(b0 + m) * DH + jj];
    int kc = jj >> 5, kk = jj & 31;
    hswz[kc][m | (((kk >> 3) & 1) << 4)][(kk & 7) | (((kk >> 4) & 1) << 3)] = (_Float16)h;
    cbuf[m][jj] = 0.0f;
  }

  v16h Breg[4][4];                               // dec_Whh^T tiles, resident across scan
#pragma unroll
  for (int j = 0; j < 4; ++j)
#pragma unroll
    for (int kc = 0; kc < 4; ++kc)
      Breg[j][kc] = *(const v16h*)(WhhB + ((((size_t)kc * 32 + (wave * 4 + j)) * 32 + lane) << 4));
  __syncthreads();

  unsigned gates_lds = (unsigned)(uintptr_t)(void*)&gates[0][0];

  for (int t = 0; t < T_OUT; ++t) {
    if (wave == 0) {                             // DMA gather of gxdecT[token] rows
      unsigned idxp[8];
#pragma unroll
      for (int p = 0; p < 8; ++p) {
        unsigned i0 = (unsigned)tgt_idx[(size_t)(b0 + 2 * p) * T_OUT + t];
        unsigned i1 = (unsigned)tgt_idx[(size_t)(b0 + 2 * p + 1) * T_OUT + t];
        idxp[p] = (i0 & 0xFFFFu) | (i1 << 16);
      }
      if (t + 1 < T_OUT) __builtin_prefetch(tgt_idx + (size_t)b0 * T_OUT + t + 1, 0, 2);
      tdm_gather16(gates_lds, gxdecT, GD, V_, idxp);
      __builtin_amdgcn_s_wait_tensorcnt(0);
    }
    __syncthreads();

    v16h a[4];
#pragma unroll
    for (int kc = 0; kc < 4; ++kc) a[kc] = *(const v16h*)(&hswz[kc][lane][0]);
    v8f acc[4] = {};
#pragma unroll
    for (int j = 0; j < 4; ++j)
#pragma unroll
      for (int kc = 0; kc < 4; ++kc) acc[j] = wmma16(a[kc], Breg[j][kc], acc[j]);

    int n = lane & 15, hi = lane >> 4;
#pragma unroll
    for (int j = 0; j < 4; ++j) {
      int g = (wave * 4 + j) * 16 + n;
#pragma unroll
      for (int r = 0; r < 8; ++r) gates[r + 8 * hi][g] += acc[j][r];
    }
    __syncthreads();

#pragma unroll
    for (int i = 0; i < 8; ++i) {
      int idx = tid + i * 256;
      int m = idx >> 7, jj = idx & 127;
      float ig = gates[m][jj], fg = gates[m][DH + jj];
      float gg = gates[m][2 * DH + jj], og = gates[m][3 * DH + jj];
      float c2 = sigmf(fg) * cbuf[m][jj] + sigmf(ig) * tanhf(gg);
      float h2 = sigmf(og) * tanhf(c2);
      cbuf[m][jj] = c2;
      int kc = jj >> 5, kk = jj & 31;
      int l2 = m | (((kk >> 3) & 1) << 4);
      int e  = (kk & 7) | (((kk >> 4) & 1) << 3);
      _Float16 hh = (_Float16)h2;
      hswz[kc][l2][e] = hh;
      dhsA[(((((size_t)t * 32 + btile) * 4 + kc) * 32 + l2) << 4) + e] = hh;   // A-swizzled dhs
    }
    __syncthreads();
  }
}

// ---------------------------------------------------------------------------
// logits[b][t][:] = dhs[t][b] @ Wd.T + f_logit[b][:]   (pure WMMA, 4096 tiles)
__global__ void k_out(const _Float16* __restrict__ dhsA, const _Float16* __restrict__ WdB,
                      const float* __restrict__ flog, float* __restrict__ out) {
  int bx = blockIdx.x;
  int t = bx >> 5, btile = bx & 31;
  int wave = threadIdx.x >> 5, lane = threadIdx.x & 31;
  v16h a[4];
#pragma unroll
  for (int kc = 0; kc < 4; ++kc)
    a[kc] = *(const v16h*)(dhsA + (((((size_t)t * 32 + btile) * 4 + kc) * 32 + lane) << 4));
  v8f acc[2] = {};
#pragma unroll
  for (int kc = 0; kc < 4; ++kc)
#pragma unroll
    for (int j = 0; j < 2; ++j) {
      int nt = wave * 2 + j;
      v16h b = *(const v16h*)(WdB + ((((size_t)kc * 16 + nt) * 32 + lane) << 4));
      acc[j] = wmma16(a[kc], b, acc[j]);
    }
  int n = lane & 15, hi = lane >> 4;
#pragma unroll
  for (int j = 0; j < 2; ++j) {
    int v = (wave * 2 + j) * 16 + n;
#pragma unroll
    for (int r = 0; r < 8; ++r) {
      int bb = btile * 16 + r + 8 * hi;
      out[((size_t)bb * T_OUT + t) * V_ + v] = acc[j][r] + flog[(size_t)bb * V_ + v];
    }
  }
}

// ---------------------------------------------------------------------------
extern "C" void kernel_launch(void* const* d_in, const int* in_sizes, int n_in,
                              void* d_out, int out_size, void* d_ws, size_t ws_size,
                              hipStream_t stream) {
  (void)in_sizes; (void)n_in; (void)out_size; (void)ws_size;
  const float* inp_onehot = (const float*)d_in[0];
  const float* zia        = (const float*)d_in[1];
  const float* tgt        = (const float*)d_in[2];
  const float* fortza     = (const float*)d_in[3];
  const float* enc_Wih    = (const float*)d_in[4];
  const float* enc_Whh    = (const float*)d_in[5];
  const float* enc_b      = (const float*)d_in[6];
  const float* sg_W       = (const float*)d_in[7];
  const float* sg_b       = (const float*)d_in[8];
  const float* Wc_W       = (const float*)d_in[9];
  const float* Wc_b       = (const float*)d_in[10];
  const float* dec_Wih    = (const float*)d_in[11];
  const float* dec_Whh    = (const float*)d_in[12];
  const float* dec_b      = (const float*)d_in[13];
  const float* out_W      = (const float*)d_in[14];
  const float* out_b      = (const float*)d_in[15];
  float* out = (float*)d_out;

  char* w = (char*)d_ws;
  size_t o = 0;
  auto alloc = [&](size_t bytes) -> char* {
    char* p = w + o;
    o += (bytes + 255) & ~(size_t)255;
    return p;
  };
  int*      inp_idx = (int*)alloc((size_t)B_ * T_IN * 4);
  int*      tgt_idx = (int*)alloc((size_t)B_ * T_OUT * 4);
  float*    WxT     = (float*)alloc((size_t)V_ * GE * 4);
  float*    gxdecT  = (float*)alloc((size_t)V_ * GD * 4);
  _Float16* ziaA    = (_Float16*)alloc((size_t)32 * 32 * 512 * 2);
  _Float16* WzB     = (_Float16*)alloc((size_t)32 * 16 * 512 * 2);
  _Float16* WhhEB   = (_Float16*)alloc((size_t)2 * 16 * 512 * 2);
  _Float16* WhhDB   = (_Float16*)alloc((size_t)4 * 32 * 512 * 2);
  _Float16* WdB     = (_Float16*)alloc((size_t)4 * 16 * 512 * 2);
  float*    gxc     = (float*)alloc((size_t)B_ * GE * 4);
  float*    h_enc   = (float*)alloc((size_t)B_ * EH * 4);
  float*    dh0     = (float*)alloc((size_t)B_ * DH * 4);
  float*    flog    = (float*)alloc((size_t)B_ * V_ * 4);
  _Float16* dhsA    = (_Float16*)alloc((size_t)T_OUT * 32 * 4 * 512 * 2);

  float* fnew = out + (size_t)B_ * T_OUT * V_;   // fortza_new tail of d_out

  // index extraction from one-hot
  k_argmax<<<(B_ * T_IN) / 8, 256, 0, stream>>>(inp_onehot, inp_idx, B_ * T_IN);
  k_argmax<<<(B_ * T_OUT) / 8, 256, 0, stream>>>(tgt, tgt_idx, B_ * T_OUT);
  // gather tables (f32)
  k_packT<<<(V_ * GE) / 256, 256, 0, stream>>>(enc_Wih, V_ + N_, nullptr, GE, WxT);
  k_packT<<<(V_ * GD) / 256, 256, 0, stream>>>(dec_Wih, V_, dec_b, GD, gxdecT);
  // WMMA operand packs (f16)
  k_packA<<<(B_ * N_) / 256, 256, 0, stream>>>(zia, B_, N_, ziaA);
  k_packB<<<(N_ * GE) / 256, 256, 0, stream>>>(enc_Wih, V_ + N_, V_, N_, GE, WzB);
  k_packB<<<(EH * GE) / 256, 256, 0, stream>>>(enc_Whh, EH, 0, EH, GE, WhhEB);
  k_packB<<<(DH * GD) / 256, 256, 0, stream>>>(dec_Whh, DH, 0, DH, GD, WhhDB);
  k_packB<<<(DH * V_) / 256, 256, 0, stream>>>(out_W, DH + SD, 0, DH, V_, WdB);
  // forward pass
  k_gxconst<<<32, 256, 0, stream>>>(ziaA, WzB, enc_b, gxc);
  k_enc_scan<<<32, 256, 0, stream>>>(inp_idx, WxT, gxc, WhhEB, h_enc);
  k_sg_fortza<<<(B_ * SD) / 256, 256, 0, stream>>>(h_enc, fortza, sg_W, sg_b, fnew);
  k_dh0<<<(B_ * DH) / 256, 256, 0, stream>>>(h_enc, zia, Wc_W, Wc_b, dh0);
  k_flogit<<<(B_ * V_) / 256, 256, 0, stream>>>(fnew, out_W, out_b, flog);
  k_dec_scan<<<32, 256, 0, stream>>>(tgt_idx, gxdecT, dh0, WhhDB, dhsA);
  k_out<<<T_OUT * 32, 256, 0, stream>>>(dhsA, WdB, flog, out);
}
